// TopDownParser_50663434224502
// MI455X (gfx1250) — compile-verified, matrix operand order checked by reading, not run on hardware
//
#include <hip/hip_runtime.h>
#include <hip/hip_bf16.h>

typedef __attribute__((ext_vector_type(16))) _Float16 v16h;
typedef __attribute__((ext_vector_type(8)))  float    v8f;

#define SEQ_T 386
#define H_DIM 256
#define EMB_D 128
#define P_DIM 385     // n+1
#define HLD   258     // padded halfs per h_lab row (conflict-free LDS, dword writes)
#define ALD   260     // padded floats per staged A row (1040 B = 16B aligned)
#define W2LD  264     // padded halfs per staged w2 row (528 B = 16B aligned)

static __device__ __forceinline__ int imin(int a, int b) { return a < b ? a : b; }

// ---------------------------------------------------------------------------
// CDNA5 async global->LDS copy (ASYNCcnt path), inline asm (portable across
// ROCm7.2 / amdgpu-toolchain). VDST = 32-bit LDS byte offset (low 32 bits of
// the generic pointer == addrspace(3) offset), VADDR = 64-bit global address.
// ---------------------------------------------------------------------------
static __device__ __forceinline__ void async_copy_b128(unsigned lds_off, const void* gaddr) {
    asm volatile("global_load_async_to_lds_b128 %0, %1, off"
                 :: "v"(lds_off), "v"(gaddr) : "memory");
}
static __device__ __forceinline__ void wait_async0() {
    asm volatile("s_wait_asynccnt 0" ::: "memory");
}
static __device__ __forceinline__ unsigned lds_off_of(const void* p) {
    return (unsigned)(size_t)p;
}

// ---------------------------------------------------------------------------
// Fragment loader: 16x32 f16 tile, row-major source (global or LDS).
// CDNA5 16-bit A/B layout: lane l holds row (l&15); VGPR v, half h ->
//   k = ((v>>2)<<4) + ((l>>4)<<3) + ((v&3)<<1) + h
// Paired halves are consecutive k -> 8 dword loads/lane (merges to b128s).
// ---------------------------------------------------------------------------
union Frag16 { v16h v; unsigned int u[8]; };

static __device__ __forceinline__ v16h load_frag_row(const _Float16* rowp, int k0, int lane) {
    Frag16 f;
    int kl = ((lane >> 4) & 1) << 3;
    #pragma unroll
    for (int v = 0; v < 8; ++v) {
        int k = k0 + ((v >> 2) << 4) + kl + ((v & 3) << 1);
        f.u[v] = *(const unsigned int*)(rowp + k);
    }
    return f.v;
}

// ---------------------------------------------------------------------------
// K1: embeddings -> f16
// ---------------------------------------------------------------------------
__global__ __launch_bounds__(128) void k_embed(const int* __restrict__ tag,
                                               const int* __restrict__ word,
                                               const float* __restrict__ te,
                                               const float* __restrict__ we,
                                               _Float16* __restrict__ emb16) {
    int t = blockIdx.x, e = threadIdx.x;
    float v = te[tag[t] * EMB_D + e] + we[word[t] * EMB_D + e];
    emb16[t * EMB_D + e] = (_Float16)v;
}

// ---------------------------------------------------------------------------
// f32 -> f16 conversion
// ---------------------------------------------------------------------------
__global__ __launch_bounds__(256) void k_cvt16(const float* __restrict__ in,
                                               _Float16* __restrict__ out, int n) {
    int i = blockIdx.x * 256 + threadIdx.x;
    if (i < n) out[i] = (_Float16)in[i];
}

// ---------------------------------------------------------------------------
// Generic small GEMM, f16 in / f32 out:  C[m][n] = sum_k A[m][k]*B[n][k] (+bias[n])
// block = 128 threads (4 waves); wave w owns mtile = blockIdx.y*4+w, ntile = blockIdx.x.
// ---------------------------------------------------------------------------
__global__ __launch_bounds__(128) void k_gemm16(const _Float16* __restrict__ A,
                                                const _Float16* __restrict__ B,
                                                const float* __restrict__ bias,
                                                float* __restrict__ C,
                                                int M, int N, int K,
                                                int lda, int ldb, int ldc) {
    int wave = threadIdx.x >> 5, lane = threadIdx.x & 31;
    int mt = blockIdx.y * 4 + wave;
    if (mt * 16 >= M) return;                     // wave-uniform exit
    int n0 = blockIdx.x * 16;

    int arow = imin(mt * 16 + (lane & 15), M - 1);  // clamp ragged M edge
    const _Float16* ap = A + (size_t)arow * lda;
    const _Float16* bp = B + (size_t)(n0 + (lane & 15)) * ldb;

    v8f c = {};
    for (int k0 = 0; k0 < K; k0 += 32) {
        v16h a = load_frag_row(ap, k0, lane);
        v16h b = load_frag_row(bp, k0, lane);
        c = __builtin_amdgcn_wmma_f32_16x16x32_f16(false, a, false, b, (short)0, c,
                                                   false, false);
    }

    int nloc = n0 + (lane & 15);
    int mb = mt * 16 + (((lane >> 4) & 1) << 3);
    float bv = bias ? bias[nloc] : 0.f;
    #pragma unroll
    for (int r = 0; r < 8; ++r) {
        int m = mb + r;
        if (m < M) C[(size_t)m * ldc + nloc] = c[r] + bv;
    }
}

// ---------------------------------------------------------------------------
// K3: sequential LSTM, one 1024-thread workgroup per direction.
// Prefetch next step's x-projection row while this step's matvec runs.
// ---------------------------------------------------------------------------
static __device__ __forceinline__ float sigm(float x) {
    return 1.f / (1.f + __expf(-x));
}

__global__ __launch_bounds__(1024) void k_lstm(const float* __restrict__ xf,
                                               const float* __restrict__ xb,
                                               const float* __restrict__ whf,
                                               const float* __restrict__ whb,
                                               float* __restrict__ hf,
                                               float* __restrict__ hb) {
    __shared__ __align__(16) float h_sh[H_DIM];
    __shared__ float g_sh[4 * H_DIM];

    int dir = blockIdx.x;
    const float* xp = dir ? xb : xf;
    const float* wh = dir ? whb : whf;
    float*       ho = dir ? hb : hf;

    int tid = threadIdx.x;
    float c = 0.f;
    if (tid < H_DIM) h_sh[tid] = 0.f;
    __syncthreads();

    const float4* w4 = (const float4*)(wh + (size_t)tid * H_DIM);
    const float4* h4 = (const float4*)h_sh;

    for (int s = 0; s < SEQ_T; ++s) {
        int t = dir ? (SEQ_T - 1 - s) : s;
        float acc = xp[(size_t)t * 1024 + tid];
        if (s + 1 < SEQ_T) {
            int tn = dir ? (t - 1) : (t + 1);
            __builtin_prefetch(xp + (size_t)tn * 1024 + tid, 0, 0);  // global_prefetch_b8
        }
        #pragma unroll 8
        for (int k = 0; k < H_DIM / 4; ++k) {
            float4 w = w4[k];
            float4 h = h4[k];
            acc += w.x * h.x + w.y * h.y + w.z * h.z + w.w * h.w;
        }
        g_sh[tid] = acc;
        __syncthreads();
        if (tid < H_DIM) {
            float gi = g_sh[tid];
            float gf = g_sh[H_DIM + tid];
            float gg = g_sh[2 * H_DIM + tid];
            float go = g_sh[3 * H_DIM + tid];
            c = sigm(gf) * c + sigm(gi) * tanhf(gg);
            float h = sigm(go) * tanhf(c);
            h_sh[tid] = h;
            ho[(size_t)t * H_DIM + tid] = h;
        }
        __syncthreads();
    }
}

// ---------------------------------------------------------------------------
// K4: build X2[t] = [ f[t], -b1[t] ] in f16   (f = hf[0:385], b1 = hb[1:386])
// ---------------------------------------------------------------------------
__global__ __launch_bounds__(512) void k_x2(const float* __restrict__ hf,
                                            const float* __restrict__ hb,
                                            _Float16* __restrict__ x2) {
    int t = blockIdx.x, k = threadIdx.x;
    float v = (k < H_DIM) ? hf[(size_t)t * H_DIM + k]
                          : -hb[(size_t)(t + 1) * H_DIM + (k - H_DIM)];
    x2[(size_t)t * 512 + k] = (_Float16)v;
}

// ---------------------------------------------------------------------------
// K5 (dominant): per 16x16 (i,j) tile:
//   stage 0: async-copy (ASYNCcnt) A-row tiles (f32) and lab_w2 (f16) into LDS
//   phase 1: h_lab = relu(Alab[j]-Alab[i]+lab_b1) -> f16 LDS (padded rows)
//   phase 2: label = h_lab @ lab_w2^T + lab_b2 via WMMA (M=256,N=128,K=256)
//   phase 3: split = dot(relu(Aspl[j]-Aspl[i]+spl_b1), spl_w2) + spl_b2 (VALU)
// LDS ~264 KB of the 320 KB WGP pool; every global byte read once per block.
// ---------------------------------------------------------------------------
__global__ __launch_bounds__(256) void k_pairs(const float* __restrict__ Alab,
                                               const float* __restrict__ Aspl,
                                               const float* __restrict__ lab_b1,
                                               const float* __restrict__ spl_b1,
                                               const _Float16* __restrict__ w2lab,
                                               const float* __restrict__ lab_b2,
                                               const _Float16* __restrict__ w2spl,
                                               const float* __restrict__ spl_b2,
                                               float* __restrict__ out) {
    __shared__ _Float16 hlab[256 * HLD];                         // 132 KB
    __shared__ __align__(16) float Ai_sh[16 * ALD];              // 16.25 KB x4
    __shared__ __align__(16) float Aj_sh[16 * ALD];
    __shared__ __align__(16) float Si_sh[16 * ALD];
    __shared__ __align__(16) float Sj_sh[16 * ALD];
    __shared__ __align__(16) _Float16 w2l_sh[128 * W2LD];        // 66 KB
    __shared__ float b1l[H_DIM], b1s[H_DIM], w2s[H_DIM], b2l[128];

    int tid = threadIdx.x;
    int it0 = blockIdx.y * 16, jt0 = blockIdx.x * 16;

    // ---- stage 0: async global->LDS staging (tracked by ASYNCcnt) ----
    {
        unsigned oAi = lds_off_of(Ai_sh), oAj = lds_off_of(Aj_sh);
        unsigned oSi = lds_off_of(Si_sh), oSj = lds_off_of(Sj_sh);
        // A tiles: 16 rows x 64 chunks(16B) each, 4 tensors
        for (int x = tid; x < 16 * 64; x += 256) {
            int r = x >> 6, ch = x & 63;
            int gi = imin(it0 + r, P_DIM - 1);
            int gj = imin(jt0 + r, P_DIM - 1);
            unsigned dro = (unsigned)(r * ALD * 4 + ch * 16);
            async_copy_b128(oAi + dro, Alab + (size_t)gi * H_DIM + ch * 4);
            async_copy_b128(oAj + dro, Alab + (size_t)gj * H_DIM + ch * 4);
            async_copy_b128(oSi + dro, Aspl + (size_t)gi * H_DIM + ch * 4);
            async_copy_b128(oSj + dro, Aspl + (size_t)gj * H_DIM + ch * 4);
        }
        // w2lab: 128 rows x 32 chunks(16B)
        unsigned oW = lds_off_of(w2l_sh);
        for (int x = tid; x < 128 * 32; x += 256) {
            int r = x >> 5, ch = x & 31;
            async_copy_b128(oW + (unsigned)(r * W2LD * 2 + ch * 16),
                            w2lab + (size_t)r * H_DIM + ch * 8);
        }
    }
    if (tid < H_DIM) {
        b1l[tid] = lab_b1[tid];
        b1s[tid] = spl_b1[tid];
        w2s[tid] = (float)w2spl[tid];
    }
    if (tid < 128) b2l[tid] = lab_b2[tid];

    int ii = tid >> 4, jj = tid & 15;
    wait_async0();
    __syncthreads();

    // ---- phase 1: hidden activations for this thread's pair (row tid) ----
    {
        const float* ai = &Ai_sh[ii * ALD];
        const float* aj = &Aj_sh[jj * ALD];
        _Float16* hr = &hlab[tid * HLD];
        #pragma unroll 4
        for (int k = 0; k < H_DIM; k += 2) {
            float h0 = fmaxf(aj[k]     - ai[k]     + b1l[k],     0.f);
            float h1 = fmaxf(aj[k + 1] - ai[k + 1] + b1l[k + 1], 0.f);
            union { unsigned int u; _Float16 h[2]; } pk;
            pk.h[0] = (_Float16)h0;
            pk.h[1] = (_Float16)h1;
            *(unsigned int*)(hr + k) = pk.u;
        }
    }
    __syncthreads();

    // ---- phase 2: label GEMM, pairs(256) x out(128) over K=256 [WMMA] ----
    {
        int wave = tid >> 5, lane = tid & 31;
        #pragma unroll
        for (int mi = 0; mi < 2; ++mi) {
            int mt = wave * 2 + mi;
            const _Float16* arow = &hlab[(mt * 16 + (lane & 15)) * HLD];
            v16h afr[8];
            #pragma unroll
            for (int kt = 0; kt < 8; ++kt)
                afr[kt] = load_frag_row(arow, kt * 32, lane);

            #pragma unroll
            for (int nt = 0; nt < 8; ++nt) {
                const _Float16* brow = &w2l_sh[(nt * 16 + (lane & 15)) * W2LD];
                v8f c = {};
                #pragma unroll
                for (int kt = 0; kt < 8; ++kt) {
                    v16h b = load_frag_row(brow, kt * 32, lane);
                    c = __builtin_amdgcn_wmma_f32_16x16x32_f16(false, afr[kt], false, b,
                                                               (short)0, c, false, false);
                }
                int n = nt * 16 + (lane & 15);
                int mb = mt * 16 + (((lane >> 4) & 1) << 3);
                float bv = b2l[n];
                #pragma unroll
                for (int r = 0; r < 8; ++r) {
                    int pr = mb + r;                     // local pair index
                    int i = it0 + (pr >> 4), j = jt0 + (pr & 15);
                    if (i < P_DIM && j < P_DIM)
                        out[((size_t)(i * P_DIM + j)) * 129 + n] = c[r] + bv;
                }
            }
        }
    }

    // ---- phase 3: split score (reads Si/Sj staged before first barrier) ----
    {
        const float* si = &Si_sh[ii * ALD];
        const float* sj = &Sj_sh[jj * ALD];
        float acc = spl_b2[0];
        #pragma unroll 4
        for (int k = 0; k < H_DIM; ++k)
            acc += fmaxf(sj[k] - si[k] + b1s[k], 0.f) * w2s[k];
        int i = it0 + ii, j = jt0 + jj;
        if (i < P_DIM && j < P_DIM)
            out[((size_t)(i * P_DIM + j)) * 129 + 128] = acc;
    }
}

// ---------------------------------------------------------------------------
// Host-side orchestration
// ---------------------------------------------------------------------------
extern "C" void kernel_launch(void* const* d_in, const int* in_sizes, int n_in,
                              void* d_out, int out_size, void* d_ws, size_t ws_size,
                              hipStream_t stream) {
    (void)in_sizes; (void)n_in; (void)out_size; (void)ws_size;

    const int*   tag_ids  = (const int*)d_in[0];
    const int*   word_ids = (const int*)d_in[1];
    const float* tag_emb  = (const float*)d_in[2];
    const float* word_emb = (const float*)d_in[3];
    const float* wxf      = (const float*)d_in[4];
    const float* whf      = (const float*)d_in[5];
    const float* bf       = (const float*)d_in[6];
    const float* wxb      = (const float*)d_in[7];
    const float* whb      = (const float*)d_in[8];
    const float* bb       = (const float*)d_in[9];
    const float* lab_w1   = (const float*)d_in[10];
    const float* lab_b1   = (const float*)d_in[11];
    const float* lab_w2   = (const float*)d_in[12];
    const float* lab_b2   = (const float*)d_in[13];
    const float* spl_w1   = (const float*)d_in[14];
    const float* spl_b1   = (const float*)d_in[15];
    const float* spl_w2   = (const float*)d_in[16];
    const float* spl_b2   = (const float*)d_in[17];
    float* out = (float*)d_out;

    // workspace carve-up (256B aligned)
    char* w = (char*)d_ws;
    size_t off = 0;
    auto carve = [&](size_t bytes) {
        void* p = w + off;
        off += (bytes + 255) & ~(size_t)255;
        return p;
    };
    _Float16* emb16  = (_Float16*)carve((size_t)SEQ_T * EMB_D * 2);
    _Float16* wxf16  = (_Float16*)carve((size_t)1024 * EMB_D * 2);
    _Float16* wxb16  = (_Float16*)carve((size_t)1024 * EMB_D * 2);
    _Float16* w1l16  = (_Float16*)carve((size_t)H_DIM * 512 * 2);
    _Float16* w1s16  = (_Float16*)carve((size_t)H_DIM * 512 * 2);
    _Float16* w2l16  = (_Float16*)carve((size_t)128 * H_DIM * 2);
    _Float16* w2s16  = (_Float16*)carve((size_t)H_DIM * 2);
    float*    xf     = (float*)carve((size_t)SEQ_T * 1024 * 4);
    float*    xb     = (float*)carve((size_t)SEQ_T * 1024 * 4);
    float*    hf     = (float*)carve((size_t)SEQ_T * H_DIM * 4);
    float*    hb     = (float*)carve((size_t)SEQ_T * H_DIM * 4);
    _Float16* x2     = (_Float16*)carve((size_t)P_DIM * 512 * 2);
    float*    Alab   = (float*)carve((size_t)P_DIM * H_DIM * 4);
    float*    Aspl   = (float*)carve((size_t)P_DIM * H_DIM * 4);

    // weight conversions to f16
    auto cvt = [&](const float* src, _Float16* dst, int n) {
        k_cvt16<<<(n + 255) / 256, 256, 0, stream>>>(src, dst, n);
    };
    cvt(wxf, wxf16, 1024 * EMB_D);
    cvt(wxb, wxb16, 1024 * EMB_D);
    cvt(lab_w1, w1l16, H_DIM * 512);
    cvt(spl_w1, w1s16, H_DIM * 512);
    cvt(lab_w2, w2l16, 128 * H_DIM);
    cvt(spl_w2, w2s16, H_DIM);

    // embeddings
    k_embed<<<SEQ_T, 128, 0, stream>>>(tag_ids, word_ids, tag_emb, word_emb, emb16);

    // x-projections: (386 x 1024) = emb16 (386x128) @ wx^T + b  [WMMA]
    k_gemm16<<<dim3(1024 / 16, (SEQ_T + 63) / 64), 128, 0, stream>>>(
        emb16, wxf16, bf, xf, SEQ_T, 1024, EMB_D, EMB_D, EMB_D, 1024);
    k_gemm16<<<dim3(1024 / 16, (SEQ_T + 63) / 64), 128, 0, stream>>>(
        emb16, wxb16, bb, xb, SEQ_T, 1024, EMB_D, EMB_D, EMB_D, 1024);

    // sequential bi-LSTM (2 workgroups, one per direction)
    k_lstm<<<2, 1024, 0, stream>>>(xf, xb, whf, whb, hf, hb);

    // X2 = [f, -b1], then A = X2 @ w1^T for both MLPs  [WMMA]
    k_x2<<<P_DIM, 512, 0, stream>>>(hf, hb, x2);
    k_gemm16<<<dim3(H_DIM / 16, (P_DIM + 63) / 64), 128, 0, stream>>>(
        x2, w1l16, nullptr, Alab, P_DIM, H_DIM, 512, 512, 512, H_DIM);
    k_gemm16<<<dim3(H_DIM / 16, (P_DIM + 63) / 64), 128, 0, stream>>>(
        x2, w1s16, nullptr, Aspl, P_DIM, H_DIM, 512, 512, 512, H_DIM);

    // dominant pair kernel: 25x25 tiles of 16x16 (i,j) pairs  [WMMA + async]
    dim3 grid((P_DIM + 15) / 16, (P_DIM + 15) / 16);
    k_pairs<<<grid, 256, 0, stream>>>(Alab, Aspl, lab_b1, spl_b1, w2l16, lab_b2,
                                      w2s16, spl_b2, out);
}
